// MultiHeadAttention_52587579572814
// MI455X (gfx1250) — matrix operand order repaired and checked by reading, not compiled, not run on hardware
//
#include <hip/hip_runtime.h>
#include <hip/hip_bf16.h>

// ---------------------------------------------------------------------------
// MultiHeadAttention forward for MI455X (gfx1250).
//   bf16 WMMA (v_wmma_f32_16x16x32_bf16) for all matmuls,
//   GLOBAL_LOAD_ASYNC_TO_LDS_B128 + double-buffered LDS for staging,
//   flash-attention (online softmax) so scores never touch HBM.
//   1/sqrt(d_k)=0.125 is a power of two -> folded exactly into the
//   Q-projection epilogue, so the flash inner loop has no score scaling.
// ---------------------------------------------------------------------------

#define BATCH   4
#define SEQ     2048
#define NHEAD   16
#define ADK     64
#define DMODEL  1024
#define MROWS   (BATCH * SEQ)   // 8192

typedef __bf16 bf16;
typedef __attribute__((ext_vector_type(16))) __bf16 v16bf;
typedef __attribute__((ext_vector_type(8)))  __bf16 bf16x8;
typedef __attribute__((ext_vector_type(8)))  float  v8f;

static __device__ __forceinline__ v16bf cat8(bf16x8 lo, bf16x8 hi) {
  return __builtin_shufflevector(lo, hi, 0,1,2,3,4,5,6,7,8,9,10,11,12,13,14,15);
}
static __device__ __forceinline__ v8f wmma_bf16(v16bf a, v16bf b, v8f c) {
  return __builtin_amdgcn_wmma_f32_16x16x32_bf16(false, a, false, b, (short)0, c,
                                                 false, false);
}
static __device__ __forceinline__ v8f v8f_zero() {
  v8f z = {0.f,0.f,0.f,0.f,0.f,0.f,0.f,0.f};
  return z;
}

// CDNA5 async copy: 16B global -> LDS, tracked by ASYNCcnt (no VGPR data path).
// VFLAT async form: VDST = LDS byte address VGPR, VADDR = 64-bit global addr.
// Generic pointers to LDS carry the LDS byte offset in their low 32 bits
// (ISA 10.2: LDS aperture access truncates to addr[31:0]).
static __device__ __forceinline__ void async_b128(unsigned lds_addr,
                                                  const void* gaddr) {
  asm volatile("global_load_async_to_lds_b128 %0, %1, off"
               :: "v"(lds_addr), "v"((unsigned long long)(uintptr_t)gaddr)
               : "memory");
}
static __device__ __forceinline__ void wait_async0() {
  asm volatile("s_wait_asynccnt 0" ::: "memory");
}
static __device__ __forceinline__ unsigned lds_off(const void* p) {
  return (unsigned)(uintptr_t)p;
}

// ---------------------------------------------------------------------------
// fp32 -> bf16 conversion (weights + activations, done once; weights are
// reused 8192x and activations 8x, so pre-converting halves L2-side traffic
// and lets every GEMM operand go through the async-to-LDS path).
// ---------------------------------------------------------------------------
__global__ void cvt_f32_to_bf16_kernel(const float* __restrict__ in,
                                       bf16* __restrict__ out, int n) {
  int i = (blockIdx.x * blockDim.x + threadIdx.x) * 4;
  if (i + 3 < n) {
    float4 f = *(const float4*)(in + i);
    out[i+0] = (bf16)f.x; out[i+1] = (bf16)f.y;
    out[i+2] = (bf16)f.z; out[i+3] = (bf16)f.w;
  } else {
    for (; i < n; ++i) out[i] = (bf16)in[i];
  }
}

// ---------------------------------------------------------------------------
// GEMM: C[M,N] = scale * (A[M,K](bf16) * W[N,K]^T(bf16) + bias[N])
//   128x128x32 tile, 256 threads = 8 waves (4 M x 2 N), 2x4 WMMA tiles/wave.
//   Double-buffered LDS, async-to-LDS staging, one barrier per K-step.
//   LDS rows padded to 40 bf16 (80B): 16B aligned, conflict-free for
//   16-consecutive-row fragment reads.
// ---------------------------------------------------------------------------
#define GBM 128
#define GBN 128
#define GBK 32
#define SPITCH 40

template<bool OUT_F32>
__global__ __launch_bounds__(256)
void gemm_xwt_kernel(const bf16* __restrict__ A,
                     const bf16* __restrict__ W,
                     const float* __restrict__ bias,
                     void* __restrict__ Out,
                     int M, int N, int K, float scale)
{
  __shared__ bf16 As[2][GBM * SPITCH];
  __shared__ bf16 Bs[2][GBN * SPITCH];

  const int tid    = threadIdx.x;
  const int lane   = tid & 31;
  const int lane16 = lane & 15;
  const int khi    = lane >> 4;
  const int wave   = tid >> 5;
  const int wm     = wave & 3;       // 4 waves x 32 rows
  const int wn     = wave >> 2;      // 2 waves x 64 cols
  const int m0     = blockIdx.x * GBM;
  const int n0     = blockIdx.y * GBN;

  const int srow   = tid >> 1;       // 0..127 staging row
  const int shalf  = tid & 1;        // 16-element half of the K slab

  auto stage = [&](int buf, int k0) {
    const bf16* asrc = A + (size_t)(m0 + srow) * K + k0 + shalf * 16;
    unsigned    adst = lds_off(&As[buf][srow * SPITCH + shalf * 16]);
    async_b128(adst,      asrc);
    async_b128(adst + 16, asrc + 8);
    const bf16* bsrc = W + (size_t)(n0 + srow) * K + k0 + shalf * 16;
    unsigned    bdst = lds_off(&Bs[buf][srow * SPITCH + shalf * 16]);
    async_b128(bdst,      bsrc);
    async_b128(bdst + 16, bsrc + 8);
  };

  v8f acc[2][4];
#pragma unroll
  for (int i = 0; i < 2; ++i)
#pragma unroll
    for (int j = 0; j < 4; ++j) acc[i][j] = v8f_zero();

  stage(0, 0);
  wait_async0();
  __syncthreads();

  const int niter = K / GBK;
  for (int it = 0; it < niter; ++it) {
    const int cur = it & 1;
    if (it + 1 < niter) stage(cur ^ 1, (it + 1) * GBK);   // prefetch next tile

    v16bf af[2], bfg[4];
#pragma unroll
    for (int tm = 0; tm < 2; ++tm) {
      const bf16* r = &As[cur][(wm*32 + tm*16 + lane16) * SPITCH];
      // A frag: halves 0..7 = K[khi*8..], halves 8..15 = K[16+khi*8..]
      af[tm] = cat8(*(const bf16x8*)&r[khi*8], *(const bf16x8*)&r[16 + khi*8]);
    }
#pragma unroll
    for (int tn = 0; tn < 4; ++tn) {
      const bf16* r = &Bs[cur][(wn*64 + tn*16 + lane16) * SPITCH];
      // B frag: halves 0..7 = K[khi*8..], halves 8..15 = K[16+khi*8..]
      bfg[tn] = cat8(*(const bf16x8*)&r[khi*8], *(const bf16x8*)&r[16 + khi*8]);
    }
#pragma unroll
    for (int tm = 0; tm < 2; ++tm)
#pragma unroll
      for (int tn = 0; tn < 4; ++tn)
        acc[tm][tn] = wmma_bf16(af[tm], bfg[tn], acc[tm][tn]);

    wait_async0();
    __syncthreads();
  }

  // epilogue: (acc + bias) * scale; store (C layout: r -> M=khi*8+r, lane16 -> N)
#pragma unroll
  for (int tn = 0; tn < 4; ++tn) {
    const int n = n0 + wn*64 + tn*16 + lane16;
    const float bv = bias[n];
#pragma unroll
    for (int tm = 0; tm < 2; ++tm) {
#pragma unroll
      for (int r = 0; r < 8; ++r) {
        const int m = m0 + wm*32 + tm*16 + khi*8 + r;
        const float val = (acc[tm][tn][r] + bv) * scale;
        if (OUT_F32) ((float*)Out)[(size_t)m * N + n] = val;
        else         ((bf16*)Out)[(size_t)m * N + n]  = (bf16)val;
      }
    }
  }
}

// ---------------------------------------------------------------------------
// Flash attention: one (b,h) 64-query tile per block; 128 threads = 4 waves,
// each wave owns 16 query rows x d=64. K tiles async-to-LDS (double buffer),
// V tiles loaded to registers early and transposed into LDS after the WMMA
// work so the wave's in-order DS queue never stalls compute.
// Q arrives pre-scaled by 1/sqrt(d_k).
// ---------------------------------------------------------------------------
#define AQ 64
#define AKT 64
#define APITCH 72

__global__ __launch_bounds__(128)
void flash_attn_kernel(const bf16* __restrict__ Q,
                       const bf16* __restrict__ Km,
                       const bf16* __restrict__ V,
                       bf16* __restrict__ O)
{
  __shared__ bf16 Qs[AQ * APITCH];
  __shared__ bf16 Ks[2][AKT * APITCH];
  __shared__ bf16 Vt[2][ADK * APITCH];      // V transposed: Vt[d][key]
  __shared__ bf16 Ps[4 * 16 * APITCH];      // per-wave P bounce (C->A relayout)

  const int tid    = threadIdx.x;
  const int lane   = tid & 31;
  const int lane16 = lane & 15;
  const int khi    = lane >> 4;
  const int wave   = tid >> 5;

  const int nqb = SEQ / AQ;                              // 32
  const int qb  = blockIdx.x % nqb;
  const int h   = (blockIdx.x / nqb) % NHEAD;
  const int b   = blockIdx.x / (nqb * NHEAD);
  const int q0  = qb * AQ;
  const size_t rowbase = ((size_t)b * SEQ) * DMODEL + (size_t)h * ADK;

  const int srow = tid >> 1, sh = tid & 1;   // staging: 2 threads/row, 32 cols
  const int vkey = tid & 63, vdh = tid >> 6; // V staging: key row, 32-d half

  // ---- stage Q tile (64x64) once, async ----
  {
    const bf16* src = Q + rowbase + (size_t)(q0 + srow) * DMODEL + sh * 32;
    unsigned    dst = lds_off(&Qs[srow * APITCH + sh * 32]);
#pragma unroll
    for (int p = 0; p < 4; ++p) async_b128(dst + 16*p, src + 8*p);
  }

  auto stageK = [&](int buf, int kt) {
    const bf16* src = Km + rowbase + (size_t)(kt + srow) * DMODEL + sh * 32;
    unsigned    dst = lds_off(&Ks[buf][srow * APITCH + sh * 32]);
#pragma unroll
    for (int p = 0; p < 4; ++p) async_b128(dst + 16*p, src + 8*p);
  };
  auto loadV = [&](int kt, bf16x8* vreg) {
    const bf16x8* src =
        (const bf16x8*)(V + rowbase + (size_t)(kt + vkey) * DMODEL + vdh * 32);
#pragma unroll
    for (int p = 0; p < 4; ++p) vreg[p] = src[p];
  };
  auto storeVt = [&](int buf, const bf16x8* vreg) {
#pragma unroll
    for (int p = 0; p < 4; ++p)
#pragma unroll
      for (int j = 0; j < 8; ++j)
        Vt[buf][(vdh*32 + p*8 + j) * APITCH + vkey] = vreg[p][j];
  };

  bf16x8 vreg[4];
  stageK(0, 0);
  loadV(0, vreg);
  storeVt(0, vreg);
  wait_async0();
  __syncthreads();   // Qs, Ks[0], Vt[0] ready

  v8f oacc[4];
#pragma unroll
  for (int tn = 0; tn < 4; ++tn) oacc[tn] = v8f_zero();
  float mrow[8], lrow[8];
#pragma unroll
  for (int r = 0; r < 8; ++r) { mrow[r] = -3.0e38f; lrow[r] = 0.0f; }

  const int nkt = SEQ / AKT;
  for (int it = 0; it < nkt; ++it) {
    const int cur = it & 1;
    const bool hasNext = (it + 1 < nkt);
    if (hasNext) {                      // prefetch next K (async) + V (regs)
      stageK(cur ^ 1, (it + 1) * AKT);
      loadV((it + 1) * AKT, vreg);
    }

    // ---- S = Qs K^T : 8 WMMAs per wave (Q pre-scaled by 1/8) ----
    v8f sc[4];
#pragma unroll
    for (int tn = 0; tn < 4; ++tn) sc[tn] = v8f_zero();
#pragma unroll
    for (int c = 0; c < 2; ++c) {
      const bf16* qr = &Qs[(wave*16 + lane16) * APITCH + c*32];
      v16bf aq = cat8(*(const bf16x8*)&qr[khi*8], *(const bf16x8*)&qr[16 + khi*8]);
#pragma unroll
      for (int tn = 0; tn < 4; ++tn) {
        const bf16* kr = &Ks[cur][(tn*16 + lane16) * APITCH + c*32];
        v16bf bk = cat8(*(const bf16x8*)&kr[khi*8], *(const bf16x8*)&kr[16 + khi*8]);
        sc[tn] = wmma_bf16(aq, bk, sc[tn]);
      }
    }

    // ---- online softmax (rows across lane16 group; 16-lane xor reduce) ----
#pragma unroll
    for (int r = 0; r < 8; ++r) {
      float s0 = sc[0][r], s1 = sc[1][r], s2 = sc[2][r], s3 = sc[3][r];
      float mx = fmaxf(fmaxf(s0, s1), fmaxf(s2, s3));
#pragma unroll
      for (int off = 1; off < 16; off <<= 1)
        mx = fmaxf(mx, __shfl_xor(mx, off, 32));
      const float mn   = fmaxf(mrow[r], mx);
      const float corr = __expf(mrow[r] - mn);
      float p0 = __expf(s0 - mn), p1 = __expf(s1 - mn);
      float p2 = __expf(s2 - mn), p3 = __expf(s3 - mn);
      sc[0][r] = p0; sc[1][r] = p1; sc[2][r] = p2; sc[3][r] = p3;
      float rs = (p0 + p1) + (p2 + p3);
#pragma unroll
      for (int off = 1; off < 16; off <<= 1)
        rs += __shfl_xor(rs, off, 32);
      lrow[r] = lrow[r] * corr + rs;
      mrow[r] = mn;
#pragma unroll
      for (int tn = 0; tn < 4; ++tn)
        oacc[tn][r] = oacc[tn][r] * corr;
    }

    // ---- bounce P through wave-private LDS to A-fragment layout ----
#pragma unroll
    for (int tn = 0; tn < 4; ++tn)
#pragma unroll
      for (int r = 0; r < 8; ++r)
        Ps[(wave*16 + khi*8 + r) * APITCH + tn*16 + lane16] = (bf16)sc[tn][r];

    // ---- O += P V : 8 WMMAs per wave ----
#pragma unroll
    for (int c = 0; c < 2; ++c) {
      const bf16* pr = &Ps[(wave*16 + lane16) * APITCH + c*32];
      v16bf ap = cat8(*(const bf16x8*)&pr[khi*8], *(const bf16x8*)&pr[16 + khi*8]);
#pragma unroll
      for (int tn = 0; tn < 4; ++tn) {
        const bf16* vr = &Vt[cur][(tn*16 + lane16) * APITCH + c*32];
        v16bf bv = cat8(*(const bf16x8*)&vr[khi*8], *(const bf16x8*)&vr[16 + khi*8]);
        oacc[tn] = wmma_bf16(ap, bv, oacc[tn]);
      }
    }

    if (hasNext) storeVt(cur ^ 1, vreg);  // deferred transpose scatter
    wait_async0();
    __syncthreads();
  }

  // ---- epilogue: divide by (l + 1e-10), store bf16 "concat" layout ----
#pragma unroll
  for (int tn = 0; tn < 4; ++tn) {
#pragma unroll
    for (int r = 0; r < 8; ++r) {
      const float inv = 1.0f / (lrow[r] + 1e-10f);
      const int s = q0 + wave*16 + khi*8 + r;
      const int d = tn*16 + lane16;
      O[rowbase + (size_t)s * DMODEL + d] = (bf16)(oacc[tn][r] * inv);
    }
  }
}

// ---------------------------------------------------------------------------
// Host side
// ---------------------------------------------------------------------------
extern "C" void kernel_launch(void* const* d_in, const int* in_sizes, int n_in,
                              void* d_out, int out_size, void* d_ws, size_t ws_size,
                              hipStream_t stream) {
  (void)in_sizes; (void)n_in; (void)out_size; (void)ws_size;

  const float* q  = (const float*)d_in[0];
  const float* k  = (const float*)d_in[1];
  const float* v  = (const float*)d_in[2];
  const float* Wq = (const float*)d_in[3];
  const float* bq = (const float*)d_in[4];
  const float* Wk = (const float*)d_in[5];
  const float* bk = (const float*)d_in[6];
  const float* Wv = (const float*)d_in[7];
  const float* bv = (const float*)d_in[8];
  const float* Wo = (const float*)d_in[9];
  const float* bo = (const float*)d_in[10];

  const size_t WELEM = (size_t)DMODEL * DMODEL;   // 1M
  const size_t TELEM = (size_t)MROWS * DMODEL;    // 8M
  bf16* Wqb = (bf16*)d_ws;
  bf16* Wkb = Wqb + WELEM;
  bf16* Wvb = Wkb + WELEM;
  bf16* Wob = Wvb + WELEM;
  bf16* qb  = Wob + WELEM;   // bf16 activations
  bf16* kb  = qb  + TELEM;
  bf16* vb  = kb  + TELEM;
  bf16* Qb  = vb  + TELEM;   // projected Q/K/V (Q pre-scaled by 1/8)
  bf16* Kb  = Qb  + TELEM;
  bf16* Vb  = Kb  + TELEM;
  bf16* Ab  = Vb  + TELEM;   // attention "concat" output

  const int cvtW = (int)(WELEM / (256 * 4));
  const int cvtT = (int)(TELEM / (256 * 4));
  cvt_f32_to_bf16_kernel<<<cvtW, 256, 0, stream>>>(Wq, Wqb, (int)WELEM);
  cvt_f32_to_bf16_kernel<<<cvtW, 256, 0, stream>>>(Wk, Wkb, (int)WELEM);
  cvt_f32_to_bf16_kernel<<<cvtW, 256, 0, stream>>>(Wv, Wvb, (int)WELEM);
  cvt_f32_to_bf16_kernel<<<cvtW, 256, 0, stream>>>(Wo, Wob, (int)WELEM);
  cvt_f32_to_bf16_kernel<<<cvtT, 256, 0, stream>>>(q,  qb,  (int)TELEM);
  cvt_f32_to_bf16_kernel<<<cvtT, 256, 0, stream>>>(k,  kb,  (int)TELEM);
  cvt_f32_to_bf16_kernel<<<cvtT, 256, 0, stream>>>(v,  vb,  (int)TELEM);

  // projections (bf16 in, bf16 out); Q gets the exact 1/sqrt(64) fold
  dim3 gg(MROWS / GBM, DMODEL / GBN);   // (64, 8)
  gemm_xwt_kernel<false><<<gg, 256, 0, stream>>>(qb, Wqb, bq, Qb, MROWS, DMODEL, DMODEL, 0.125f);
  gemm_xwt_kernel<false><<<gg, 256, 0, stream>>>(kb, Wkb, bk, Kb, MROWS, DMODEL, DMODEL, 1.0f);
  gemm_xwt_kernel<false><<<gg, 256, 0, stream>>>(vb, Wvb, bv, Vb, MROWS, DMODEL, DMODEL, 1.0f);

  // flash attention: one block per (b, h, 64-query tile)
  const int ablocks = BATCH * NHEAD * (SEQ / AQ);  // 2048
  flash_attn_kernel<<<ablocks, 128, 0, stream>>>(Qb, Kb, Vb, Ab);

  // output projection (bf16 in, fp32 out)
  gemm_xwt_kernel<true><<<gg, 256, 0, stream>>>(Ab, Wob, bo, d_out, MROWS, DMODEL, DMODEL, 1.0f);
}
